// SAB_57131654971491
// MI455X (gfx1250) — matrix operand order repaired and checked by reading, not compiled, not run on hardware
//
#include <hip/hip_runtime.h>
#include <hip/hip_bf16.h>

// ---------------------------------------------------------------------------
// Problem constants (reference: B=4, S=2048, H=1024, NHEADS=16, dh=64)
// ---------------------------------------------------------------------------
constexpr int  kBN  = 4;
constexpr int  kS   = 2048;
constexpr int  kH   = 1024;
constexpr int  kNH  = 16;
constexpr int  kDH  = 64;
constexpr float kEps = 1e-5f;
// exponent scale: softmax uses exp(s/8) = exp2(s * 0.125 * log2(e))
constexpr float kC1 = 0.125f * 1.4426950408889634f;

typedef __attribute__((ext_vector_type(16))) __bf16 v16bf;
typedef __attribute__((ext_vector_type(8)))  __bf16 v8bf;
typedef __attribute__((ext_vector_type(8)))  float  v8f;

#define WMMA_BF16(A, B, C) \
  __builtin_amdgcn_wmma_f32_16x16x32_bf16(false, (A), false, (B), (short)0, (C), false, false)

// ds_swizzle_b32 xor-swap within 32 lanes: offset = {0, xor[14:10], or[9:5], and[4:0]}
#define SWZ_XOR(x, MASK) \
  __builtin_bit_cast(float, __builtin_amdgcn_ds_swizzle(__builtin_bit_cast(int, (x)), \
                                                        0x1f | ((MASK) << 10)))

// A-matrix (16x32 bf16) per-lane fragment from a bf16 row:
//   elems 0..7 at p[lo..lo+7], elems 8..15 at p[hi2..hi2+7]
__device__ __forceinline__ v16bf load_a_bf16(const __bf16* __restrict__ p, int lo, int hi2) {
  v8bf alo = *(const v8bf*)(p + lo);
  v8bf ahi = *(const v8bf*)(p + hi2);
  return __builtin_shufflevector(alo, ahi, 0, 1, 2, 3, 4, 5, 6, 7,
                                 8, 9, 10, 11, 12, 13, 14, 15);
}

// ---------------------------------------------------------------------------
// Kernel 1: f32 -> bf16 weight conversion (W reused by every GEMM tile)
// ---------------------------------------------------------------------------
__global__ __launch_bounds__(256) void cvt_w_kernel(const float* __restrict__ w,
                                                    __bf16* __restrict__ wb, int n) {
  for (int i = blockIdx.x * 256 + threadIdx.x; i < n; i += gridDim.x * 256)
    wb[i] = (__bf16)w[i];
}

// ---------------------------------------------------------------------------
// Kernel 2: q prep. Emits qbf (bf16, [B,S,H]) and per-head transposed
// qbt (bf16, [B*NH, DH, S]) via an LDS 64x65 tile (both sides coalesced).
// ---------------------------------------------------------------------------
__global__ __launch_bounds__(256) void qprep_kernel(const float* __restrict__ q,
                                                    __bf16* __restrict__ qbf,
                                                    __bf16* __restrict__ qbt) {
  __shared__ __bf16 tile[64 * 65];
  const int bid = blockIdx.x;
  const int sb  = bid & 31;                  // S/64 = 32 s-blocks
  const int h   = (bid >> 5) & 15;
  const int b   = bid >> 9;
  const int s0  = sb * 64;
  const int bh  = b * kNH + h;
#pragma unroll
  for (int i = 0; i < 16; ++i) {
    const int idx = i * 256 + threadIdx.x;
    const int sl = idx >> 6, d = idx & 63;
    const size_t off = ((size_t)b * kS + s0 + sl) * kH + h * kDH + d;
    __bf16 bv = (__bf16)q[off];
    qbf[off] = bv;
    tile[d * 65 + sl] = bv;
  }
  __syncthreads();
#pragma unroll
  for (int i = 0; i < 16; ++i) {
    const int idx = i * 256 + threadIdx.x;
    const int d = idx >> 6, sl = idx & 63;
    qbt[((size_t)bh * kDH + d) * kS + s0 + sl] = tile[d * 65 + sl];
  }
}

// ---------------------------------------------------------------------------
// Kernel 3: fused attention. One WG (4 waves, 128 thr) per (b, h, 16-row qtile).
// Each wave owns 512 key columns. Three-pass softmax (max / sumexp / write):
// extra co-executing WMMA traded for fewer VALU+transcendental ops. All matrix
// operands are preconverted bf16 (contiguous v16bf loads, zero cvt in loops).
// Writes attn (f32, non-temporal) and y0 = q + ctx.
// ---------------------------------------------------------------------------
__global__ __launch_bounds__(128) void attn_kernel(const __bf16* __restrict__ qbf,
                                                   const __bf16* __restrict__ qbt,
                                                   const float* __restrict__ q,
                                                   float* __restrict__ attn_out,
                                                   float* __restrict__ y0) {
  __shared__ __bf16 Pstage[4][16 * 32];       // per-wave P tile staging (bf16)
  __shared__ float  mshare[4][16], lshare[4][16];
  __shared__ float  ctxsh[4][16][kDH];        // per-wave partial ctx

  const int lane = threadIdx.x & 31;
  const int w    = threadIdx.x >> 5;
  const int m16  = lane & 15;
  const int hi   = lane >> 4;

  const int bid   = blockIdx.x;
  const int qt    = bid & 127;                // S/16 = 128 q tiles
  const int h     = (bid >> 7) & 15;
  const int b     = bid >> 11;
  const int qrow0 = qt * 16;
  const int hcol0 = h * kDH;
  const int bh    = b * kNH + h;
  const __bf16* qbb = qbf + (size_t)b * kS * kH;

  // ---- A tile: 16 query rows x 64 (dh), two K-steps of 32, kept in regs ----
  const __bf16* arow = qbb + (size_t)(qrow0 + m16) * kH + hcol0;
  v16bf aq0 = load_a_bf16(arow, hi * 16, hi * 16 + 8);
  v16bf aq1 = load_a_bf16(arow + 32, hi * 16, hi * 16 + 8);

  const int cg0 = w * 32;                     // this wave: col groups [cg0, cg0+32)

  // ---------------- Pass 1: row max (scale folded out; scale > 0) ----------
  float mr[8];
#pragma unroll
  for (int r = 0; r < 8; ++r) mr[r] = -1e30f;

  for (int cg = cg0; cg < cg0 + 32; ++cg) {
    const __bf16* brow = qbb + (size_t)(cg * 16 + m16) * kH + hcol0 + hi * 16;
    __builtin_prefetch(brow + 16 * kH, 0, 1);          // next key tile
    v16bf b0 = *(const v16bf*)brow;
    v16bf b1 = *(const v16bf*)(brow + 32);
    v8f d = {};
    d = WMMA_BF16(aq0, b0, d);
    d = WMMA_BF16(aq1, b1, d);
#pragma unroll
    for (int r = 0; r < 8; ++r) mr[r] = fmaxf(mr[r], d[r]);
  }
#pragma unroll
  for (int r = 0; r < 8; ++r) {
    mr[r] = fmaxf(mr[r], SWZ_XOR(mr[r], 8));
    mr[r] = fmaxf(mr[r], SWZ_XOR(mr[r], 4));
    mr[r] = fmaxf(mr[r], SWZ_XOR(mr[r], 2));
    mr[r] = fmaxf(mr[r], SWZ_XOR(mr[r], 1));
  }
  if (m16 == 0) {
#pragma unroll
    for (int r = 0; r < 8; ++r) mshare[w][hi * 8 + r] = mr[r];
  }
  __syncthreads();
  float nm2[8];                                // -max*c1 (exp2 domain)
#pragma unroll
  for (int r = 0; r < 8; ++r) {
    const int row = hi * 8 + r;
    float m = fmaxf(fmaxf(mshare[0][row], mshare[1][row]),
                    fmaxf(mshare[2][row], mshare[3][row]));
    nm2[r] = -m * kC1;
  }

  // ---------------- Pass 2: row sum of exp2(d*c1 - m*c1) -------------------
  float lr[8];
#pragma unroll
  for (int r = 0; r < 8; ++r) lr[r] = 0.f;

  for (int cg = cg0; cg < cg0 + 32; ++cg) {
    const __bf16* brow = qbb + (size_t)(cg * 16 + m16) * kH + hcol0 + hi * 16;
    v16bf b0 = *(const v16bf*)brow;
    v16bf b1 = *(const v16bf*)(brow + 32);
    v8f d = {};
    d = WMMA_BF16(aq0, b0, d);
    d = WMMA_BF16(aq1, b1, d);
#pragma unroll
    for (int r = 0; r < 8; ++r)
      lr[r] += __builtin_amdgcn_exp2f(fmaf(d[r], kC1, nm2[r]));
  }
#pragma unroll
  for (int r = 0; r < 8; ++r) {
    lr[r] += SWZ_XOR(lr[r], 8);
    lr[r] += SWZ_XOR(lr[r], 4);
    lr[r] += SWZ_XOR(lr[r], 2);
    lr[r] += SWZ_XOR(lr[r], 1);
  }
  if (m16 == 0) {
#pragma unroll
    for (int r = 0; r < 8; ++r) lshare[w][hi * 8 + r] = lr[r];
  }
  __syncthreads();
  float c3[8];                                 // -m*c1 - log2(l): fold 1/l into exp2
#pragma unroll
  for (int r = 0; r < 8; ++r) {
    const int row = hi * 8 + r;
    float l = lshare[0][row] + lshare[1][row] + lshare[2][row] + lshare[3][row];
    c3[r] = nm2[r] - __builtin_amdgcn_logf(l);
  }

  // ---------------- Pass 3: write attn, accumulate ctx = P*K ----------------
  v8f acc[4] = {};
  float* attnBase = attn_out + (((size_t)bh) * kS + qrow0) * kS;
  __bf16* ps = Pstage[w];
  const __bf16* qtb = qbt + (size_t)bh * kDH * kS;

  for (int ch = 0; ch < 16; ++ch) {           // 16 chunks of 32 key cols
#pragma unroll
    for (int sub = 0; sub < 2; ++sub) {
      const int cg = cg0 + ch * 2 + sub;
      const int kc = cg * 16;
      const __bf16* brow = qbb + (size_t)(kc + m16) * kH + hcol0 + hi * 16;
      v16bf b0 = *(const v16bf*)brow;
      v16bf b1 = *(const v16bf*)(brow + 32);
      v8f d = {};
      d = WMMA_BF16(aq0, b0, d);
      d = WMMA_BF16(aq1, b1, d);
#pragma unroll
      for (int r = 0; r < 8; ++r) {
        const int row = hi * 8 + r;
        float p = __builtin_amdgcn_exp2f(fmaf(d[r], kC1, c3[r]));  // normalized
        __builtin_nontemporal_store(p, attnBase + (size_t)row * kS + kc + m16);
        ps[row * 32 + sub * 16 + m16] = (__bf16)p;  // stage for 2nd matmul A
      }
    }
    // A2 = P tile (16 x 32) from LDS, contiguous k = 16*hi + s per lane (same order as B2)
    v16bf a2 = load_a_bf16(ps + m16 * 32, hi * 16, hi * 16 + 8);
    const int kcc = (cg0 + ch * 2) * 16;      // chunk's first key col
#pragma unroll
    for (int dg = 0; dg < 4; ++dg) {          // 4 dh groups of 16
      // B2(k,n) = K[kcc+k][dg*16+n] = qbt[dg*16+n][kcc+k]: contiguous per lane
      const __bf16* btrow = qtb + (size_t)(dg * 16 + m16) * kS + kcc + hi * 16;
      v16bf bt = *(const v16bf*)btrow;
      acc[dg] = WMMA_BF16(a2, bt, acc[dg]);
    }
  }

  // ---------------- Reduce ctx across waves, add residual ----------------
#pragma unroll
  for (int dg = 0; dg < 4; ++dg)
#pragma unroll
    for (int r = 0; r < 8; ++r)
      ctxsh[w][hi * 8 + r][dg * 16 + m16] = acc[dg][r];
  __syncthreads();
  const float* qf = q + (size_t)b * kS * kH;
  for (int idx = threadIdx.x; idx < 16 * kDH; idx += 128) {
    const int row = idx >> 6, d = idx & 63;
    float s = ctxsh[0][row][d] + ctxsh[1][row][d] + ctxsh[2][row][d] + ctxsh[3][row][d];
    s += qf[(size_t)(qrow0 + row) * kH + hcol0 + d];
    y0[((size_t)b * kS + qrow0 + row) * kH + hcol0 + d] = s;
  }
}

// ---------------------------------------------------------------------------
// Kernel 4/6: LayerNorm over H=1024, one block per row. Optionally also emits
// a bf16 copy (consumed by the WMMA GEMM).
// ---------------------------------------------------------------------------
__global__ __launch_bounds__(256) void ln_kernel(const float* __restrict__ x,
                                                 const float* __restrict__ g,
                                                 const float* __restrict__ bta,
                                                 float* __restrict__ yf,
                                                 __bf16* __restrict__ yb) {
  __shared__ float red[256];
  const int row = blockIdx.x;
  const float* xr = x + (size_t)row * kH;
  float v[4];
  float s = 0.f;
#pragma unroll
  for (int i = 0; i < 4; ++i) { v[i] = xr[threadIdx.x + i * 256]; s += v[i]; }
  red[threadIdx.x] = s;
  __syncthreads();
  for (int o = 128; o > 0; o >>= 1) {
    if (threadIdx.x < o) red[threadIdx.x] += red[threadIdx.x + o];
    __syncthreads();
  }
  const float mu = red[0] * (1.f / kH);
  __syncthreads();
  float s2 = 0.f;
#pragma unroll
  for (int i = 0; i < 4; ++i) { float d = v[i] - mu; s2 += d * d; }
  red[threadIdx.x] = s2;
  __syncthreads();
  for (int o = 128; o > 0; o >>= 1) {
    if (threadIdx.x < o) red[threadIdx.x] += red[threadIdx.x + o];
    __syncthreads();
  }
  const float rstd = rsqrtf(red[0] * (1.f / kH) + kEps);
#pragma unroll
  for (int i = 0; i < 4; ++i) {
    const int c = threadIdx.x + i * 256;
    float o = (v[i] - mu) * rstd * g[c] + bta[c];
    yf[(size_t)row * kH + c] = o;
    if (yb) yb[(size_t)row * kH + c] = (__bf16)o;
  }
}

// ---------------------------------------------------------------------------
// Kernel 5: out2 = x + relu(x @ W^T + b). bf16 WMMA GEMM, one wave per
// 32x64 output tile (2 M-tiles share each B tile). M=8192, N=K=1024.
// ---------------------------------------------------------------------------
__global__ __launch_bounds__(256) void mlp_kernel(const __bf16* __restrict__ xb,
                                                  const __bf16* __restrict__ wb,
                                                  const float* __restrict__ bias,
                                                  const float* __restrict__ xf,
                                                  float* __restrict__ out2) {
  const int lane = threadIdx.x & 31;
  const int gw   = blockIdx.x * 8 + (threadIdx.x >> 5);
  const int r0   = (gw & 255) * 32;           // 8192/32 = 256 row tiles
  const int c0   = (gw >> 8) * 64;            // 1024/64 = 16 col tiles
  const int m16  = lane & 15;
  const int hi   = lane >> 4;

  v8f acc[2][4] = {};
  const __bf16* arow0 = xb + (size_t)(r0 + m16) * kH;
  const __bf16* arow1 = arow0 + (size_t)16 * kH;
  for (int k0 = 0; k0 < kH; k0 += 32) {
    v16bf a0 = load_a_bf16(arow0 + k0, hi * 16, hi * 16 + 8);
    v16bf a1 = load_a_bf16(arow1 + k0, hi * 16, hi * 16 + 8);
#pragma unroll
    for (int ng = 0; ng < 4; ++ng) {
      // B(k,n) = W[c0+ng*16+n][k]: 16 contiguous bf16 per lane, shared by both M-tiles
      const __bf16* brow = wb + (size_t)(c0 + ng * 16 + m16) * kH + k0 + hi * 16;
      v16bf bm = *(const v16bf*)brow;
      acc[0][ng] = WMMA_BF16(a0, bm, acc[0][ng]);
      acc[1][ng] = WMMA_BF16(a1, bm, acc[1][ng]);
    }
  }
#pragma unroll
  for (int mt = 0; mt < 2; ++mt) {
#pragma unroll
    for (int ng = 0; ng < 4; ++ng) {
      const int col = c0 + ng * 16 + m16;
      const float bv = bias[col];
#pragma unroll
      for (int r = 0; r < 8; ++r) {
        const int row = r0 + mt * 16 + r + hi * 8;
        float z = fmaxf(acc[mt][ng][r] + bv, 0.f);
        out2[(size_t)row * kH + col] = xf[(size_t)row * kH + col] + z;
      }
    }
  }
}

// ---------------------------------------------------------------------------
// Host launcher
// ---------------------------------------------------------------------------
extern "C" void kernel_launch(void* const* d_in, const int* in_sizes, int n_in,
                              void* d_out, int out_size, void* d_ws, size_t ws_size,
                              hipStream_t stream) {
  const float* query = (const float*)d_in[0];
  const float* out_w = (const float*)d_in[1];
  const float* out_b = (const float*)d_in[2];
  const float* ln0_g = (const float*)d_in[3];
  const float* ln0_b = (const float*)d_in[4];
  const float* ln1_g = (const float*)d_in[5];
  const float* ln1_b = (const float*)d_in[6];

  float* out  = (float*)d_out;                                  // [B,S,H]
  float* attn = out + (size_t)kBN * kS * kH;                    // [B,NH,S,S]

  char* ws = (char*)d_ws;
  float*  y0  = (float*)ws;                           // 32 MB, reused as out2
  float*  x1  = (float*)(ws + (size_t)33554432);      // 32 MB
  __bf16* x1b = (__bf16*)(ws + (size_t)67108864);     // 16 MB
  __bf16* wbf = (__bf16*)(ws + (size_t)83886080);     //  2 MB
  __bf16* qbf = (__bf16*)(ws + (size_t)85983232);     // 16 MB
  __bf16* qbt = (__bf16*)(ws + (size_t)102760448);    // 16 MB

  const int rows = kBN * kS;                          // 8192

  // 1) W -> bf16
  cvt_w_kernel<<<1024, 256, 0, stream>>>(out_w, wbf, kH * kH);
  // 2) q -> bf16 (row-major) + per-head transposed bf16
  qprep_kernel<<<kBN * kNH * (kS / 64), 256, 0, stream>>>(query, qbf, qbt);
  // 3) attention: attn + y0 = q + ctx
  attn_kernel<<<kBN * kNH * (kS / 16), 128, 0, stream>>>(qbf, qbt, query, attn, y0);
  // 4) LN0 -> x1 (f32) + x1b (bf16)
  ln_kernel<<<rows, 256, 0, stream>>>(y0, ln0_g, ln0_b, x1, x1b);
  // 5) MLP: out2 (aliases y0) = x1 + relu(x1 @ W^T + b)
  mlp_kernel<<<(rows / 32) * (kH / 64) / 8, 256, 0, stream>>>(x1b, wbf, out_b, x1, y0);
  // 6) LN1 -> final out
  ln_kernel<<<rows, 256, 0, stream>>>(y0, ln1_g, ln1_b, out, nullptr);
}
